// Attention_24507083391523
// MI455X (gfx1250) — matrix operand order repaired and checked by reading, not compiled, not run on hardware
//
#include <hip/hip_runtime.h>
#include <hip/hip_bf16.h>
#include <math.h>

typedef _Float16 f16;
typedef __attribute__((ext_vector_type(16))) _Float16 v16h;
typedef __attribute__((ext_vector_type(8)))  _Float16 v8h;
typedef __attribute__((ext_vector_type(4)))  _Float16 v4h;
typedef __attribute__((ext_vector_type(8)))  float    v8f;
typedef __attribute__((ext_vector_type(4)))  float    v4f;

#define WMMA_F16(a,b,c) __builtin_amdgcn_wmma_f32_16x16x32_f16(false,(a),false,(b),(short)0,(c),false,false)

constexpr int Bc  = 2;
constexpr int Sc  = 2048;
constexpr int Dc  = 2048;
constexpr int Hc  = 16;
constexpr int HDc = 128;

__device__ __forceinline__ v16h join16(v8h a, v8h b) {
  v16h r;
#pragma unroll
  for (int i = 0; i < 8; ++i) { r[i] = a[i]; r[i + 8] = b[i]; }
  return r;
}

// CDNA5 async memory->LDS copy (16B per lane), tracked by ASYNCcnt.
__device__ __forceinline__ void async_ld_b128(unsigned lds_addr, const f16* gptr) {
  unsigned long long ga = (unsigned long long)(uintptr_t)gptr;
  asm volatile("global_load_async_to_lds_b128 %0, %1, off"
               :: "v"(lds_addr), "v"(ga) : "memory");
}
__device__ __forceinline__ void wait_async0() {
  asm volatile("s_wait_asynccnt 0" ::: "memory");
}
// low 32 bits of a generic LDS pointer == LDS byte offset
__device__ __forceinline__ unsigned lds_off(const void* p) {
  return (unsigned)(uintptr_t)p;
}

// ---------------------------------------------------------------------------
// fp32 -> fp16 conversion, 4 elements/thread
// ---------------------------------------------------------------------------
__global__ void f32_to_f16_kernel(const float* __restrict__ in, f16* __restrict__ out, int n4) {
  int i = blockIdx.x * 256 + threadIdx.x;
  if (i < n4) {
    v4f v = ((const v4f*)in)[i];
    v4h o;
#pragma unroll
    for (int k = 0; k < 4; ++k) o[k] = (f16)v[k];
    ((v4h*)out)[i] = o;
  }
}

// ---------------------------------------------------------------------------
// GEMM: C[m,n] = sum_k A[m,k] * W[n,k]   (A: MxK row-major f16, W: NxK row-major f16)
// Block tile 128x128, BK=32, 256 threads = 8 waves (2x4), wave tile 64x32.
// Tiles staged via CDNA5 async global->LDS copies, double-buffered.
// ---------------------------------------------------------------------------
template <typename OutT>
__global__ __launch_bounds__(256) void gemm_xwT_kernel(const f16* __restrict__ A,
                                                       const f16* __restrict__ W,
                                                       OutT* __restrict__ C,
                                                       int M, int N, int K) {
  __shared__ __align__(16) f16 As[2][128][40];
  __shared__ __align__(16) f16 Ws[2][128][40];

  const int NB = N / 128;
  const int bm = blockIdx.x / NB;
  const int bn = blockIdx.x % NB;
  const int m0 = bm * 128, n0 = bn * 128;

  const int tid  = threadIdx.x;
  const int w    = tid >> 5;
  const int lane = tid & 31;
  const int lo   = lane & 15;
  const int hi   = lane >> 4;
  const int wr   = w >> 2;   // 0..1 : wave row (64 rows each)
  const int wc   = w & 3;    // 0..3 : wave col (32 cols each)

  v8f acc[4][2];
#pragma unroll
  for (int i = 0; i < 4; ++i)
#pragma unroll
    for (int j = 0; j < 2; ++j)
#pragma unroll
      for (int r = 0; r < 8; ++r) acc[i][j][r] = 0.0f;

  const int lr = tid >> 1;          // 0..127
  const int lc = (tid & 1) * 16;    // 0 or 16
  const f16* pa0 = A + (size_t)(m0 + lr) * K + lc;
  const f16* pw0 = W + (size_t)(n0 + lr) * K + lc;

  // stage tile at k0 into buffer p (4 async 16B transfers per thread)
  auto stage = [&](int p, int k0) {
    unsigned la = lds_off(&As[p][lr][lc]);
    unsigned lw = lds_off(&Ws[p][lr][lc]);
    async_ld_b128(la,      pa0 + k0);
    async_ld_b128(la + 16, pa0 + k0 + 8);
    async_ld_b128(lw,      pw0 + k0);
    async_ld_b128(lw + 16, pw0 + k0 + 8);
  };

  stage(0, 0);
  wait_async0();
  __syncthreads();

  int p = 0;
  for (int k0 = 0; k0 < K; k0 += 32) {
    if (k0 + 32 < K) stage(p ^ 1, k0 + 32);

    v16h af[4];
#pragma unroll
    for (int i = 0; i < 4; ++i) {
      int row = wr * 64 + i * 16 + lo;
      af[i] = join16(*(const v8h*)&As[p][row][hi * 8],
                     *(const v8h*)&As[p][row][16 + hi * 8]);
    }
    v16h bf[2];
#pragma unroll
    for (int j = 0; j < 2; ++j) {
      int col = wc * 32 + j * 16 + lo;
      bf[j] = join16(*(const v8h*)&Ws[p][col][hi * 8],
                     *(const v8h*)&Ws[p][col][16 + hi * 8]);
    }
#pragma unroll
    for (int i = 0; i < 4; ++i)
#pragma unroll
      for (int j = 0; j < 2; ++j)
        acc[i][j] = WMMA_F16(af[i], bf[j], acc[i][j]);

    wait_async0();
    __syncthreads();
    p ^= 1;
  }

#pragma unroll
  for (int i = 0; i < 4; ++i)
#pragma unroll
    for (int j = 0; j < 2; ++j)
#pragma unroll
      for (int r = 0; r < 8; ++r) {
        int row = m0 + wr * 64 + i * 16 + r + 8 * hi;
        int col = n0 + wc * 32 + j * 16 + lo;
        C[(size_t)row * N + col] = (OutT)acc[i][j][r];
      }
}

// ---------------------------------------------------------------------------
// RoPE in-place on f16 [b,s,h,hd]; interleaved even/odd pairs.
// grid covers exactly B*S*H*(HD/2) threads.
// ---------------------------------------------------------------------------
__global__ void rope_kernel(f16* __restrict__ t, const float* __restrict__ ct,
                            const float* __restrict__ st) {
  int i = blockIdx.x * 256 + threadIdx.x;       // ((b*S+s)*H + h)*64 + p
  int p  = i & 63;
  int h  = (i >> 6) & (Hc - 1);
  int bs = i >> 10;                              // b*S + s
  int s  = bs & (Sc - 1);
  float c  = ct[s * 64 + p];
  float sn = st[s * 64 + p];
  size_t base = (size_t)bs * Dc + h * HDc + 2 * p;
  float e = (float)t[base];
  float o = (float)t[base + 1];
  t[base]     = (f16)(e * c - o * sn);
  t[base + 1] = (f16)(e * sn + o * c);
}

// ---------------------------------------------------------------------------
// V transpose: vh[b,s,h,d] -> vt[b,h,d,s], 64x64 LDS tiles.
// ---------------------------------------------------------------------------
__global__ __launch_bounds__(256) void vtrans_kernel(const f16* __restrict__ vh,
                                                     f16* __restrict__ vt) {
  __shared__ __align__(16) f16 tile[64][72];
  int bh = blockIdx.z;
  int b = bh / Hc, h = bh % Hc;
  int s0 = blockIdx.y * 64;
  int d0 = blockIdx.x * 64;
  int tid = threadIdx.x;
#pragma unroll
  for (int it = 0; it < 4; ++it) {
    int idx = it * 256 + tid;
    int sr  = idx >> 4;
    int dd  = (idx & 15) * 4;
    const f16* p = vh + ((size_t)(b * Sc + s0 + sr) * Hc + h) * HDc + d0 + dd;
    *(v4h*)&tile[sr][dd] = *(const v4h*)p;
  }
  __syncthreads();
#pragma unroll
  for (int it = 0; it < 4; ++it) {
    int idx = it * 256 + tid;
    int dr  = idx >> 4;
    int sc  = (idx & 15) * 4;
    v4h val;
#pragma unroll
    for (int k = 0; k < 4; ++k) val[k] = tile[sc + k][dr];
    *(v4h*)(vt + (((size_t)b * Hc + h) * HDc + d0 + dr) * Sc + s0 + sc) = val;
  }
}

// ---------------------------------------------------------------------------
// Flash attention (causal). 128 threads = 4 waves; each wave owns 16 q-rows.
// qh/kh: f16 [b,s,h,d]; vt: f16 [b,h,d,s]; ao: f16 [b,s,h*d].
// ---------------------------------------------------------------------------
__global__ __launch_bounds__(128) void attn_kernel(const f16* __restrict__ qh,
                                                   const f16* __restrict__ kh,
                                                   const f16* __restrict__ vt,
                                                   f16* __restrict__ ao) {
  __shared__ __align__(16) f16 Ps[4][16][40];
  constexpr int QT = Sc / 64;
  int blk = blockIdx.x;
  int qt = blk % QT;
  int h  = (blk / QT) % Hc;
  int b  = blk / (QT * Hc);

  const int w    = threadIdx.x >> 5;
  const int lane = threadIdx.x & 31;
  const int lo   = lane & 15;
  const int hi   = lane >> 4;
  const int qb   = qt * 64 + w * 16;

  // Q fragments: 4 d-chunks of 32, A-layout.
  v16h qf[4];
  const f16* qrow = qh + ((size_t)(b * Sc + qb + lo) * Hc + h) * HDc;
#pragma unroll
  for (int c = 0; c < 4; ++c)
    qf[c] = join16(*(const v8h*)(qrow + c * 32 + hi * 8),
                   *(const v8h*)(qrow + c * 32 + 16 + hi * 8));

  v8f o[8];
  float m[8], l[8];
#pragma unroll
  for (int jj = 0; jj < 8; ++jj)
#pragma unroll
    for (int r = 0; r < 8; ++r) o[jj][r] = 0.0f;
#pragma unroll
  for (int r = 0; r < 8; ++r) { m[r] = -1e30f; l[r] = 0.0f; }

  const float scale = 0.088388347648318447f;  // 1/sqrt(128)
  const f16* kbase = kh + ((size_t)b * Sc * Hc + h) * HDc;
  const f16* vbase = vt + ((size_t)b * Hc + h) * HDc * (size_t)Sc;

  const int nkt = (qb + 15) / 32 + 1;
  for (int kt = 0; kt < nkt; ++kt) {
    const int k0 = kt * 32;
    v8f sc2[2];
#pragma unroll
    for (int j = 0; j < 2; ++j) {
#pragma unroll
      for (int r = 0; r < 8; ++r) sc2[j][r] = 0.0f;
      const f16* krow = kbase + (size_t)(k0 + j * 16 + lo) * Dc;
#pragma unroll
      for (int c = 0; c < 4; ++c) {
        v16h kf = join16(*(const v8h*)(krow + c * 32 + hi * 8),
                         *(const v8h*)(krow + c * 32 + 16 + hi * 8));
        sc2[j] = WMMA_F16(qf[c], kf, sc2[j]);
      }
    }
    // scale + causal mask (only boundary tiles can mask)
    const bool boundary = (k0 + 31 > qb);
#pragma unroll
    for (int j = 0; j < 2; ++j)
#pragma unroll
      for (int r = 0; r < 8; ++r) {
        float v = sc2[j][r] * scale;
        if (boundary) {
          int qpos = qb + r + 8 * hi;
          int kpos = k0 + j * 16 + lo;
          if (kpos > qpos) v = -1e30f;
        }
        sc2[j][r] = v;
      }
    // row max over 16-lane halves
    float mt[8];
#pragma unroll
    for (int r = 0; r < 8; ++r) mt[r] = fmaxf(sc2[0][r], sc2[1][r]);
#pragma unroll
    for (int msk = 1; msk < 16; msk <<= 1)
#pragma unroll
      for (int r = 0; r < 8; ++r) mt[r] = fmaxf(mt[r], __shfl_xor(mt[r], msk, 32));

    float alpha[8];
#pragma unroll
    for (int r = 0; r < 8; ++r) {
      float mn = fmaxf(m[r], mt[r]);
      alpha[r] = __expf(m[r] - mn);
      m[r] = mn;
    }
#pragma unroll
    for (int j = 0; j < 2; ++j)
#pragma unroll
      for (int r = 0; r < 8; ++r) sc2[j][r] = __expf(sc2[j][r] - m[r]);

    float rs[8];
#pragma unroll
    for (int r = 0; r < 8; ++r) rs[r] = sc2[0][r] + sc2[1][r];
#pragma unroll
    for (int msk = 1; msk < 16; msk <<= 1)
#pragma unroll
      for (int r = 0; r < 8; ++r) rs[r] += __shfl_xor(rs[r], msk, 32);
#pragma unroll
    for (int r = 0; r < 8; ++r) l[r] = l[r] * alpha[r] + rs[r];
#pragma unroll
    for (int jj = 0; jj < 8; ++jj)
#pragma unroll
      for (int r = 0; r < 8; ++r) o[jj][r] *= alpha[r];

    // C-layout -> A-layout via per-wave LDS scratch
#pragma unroll
    for (int j = 0; j < 2; ++j)
#pragma unroll
      for (int r = 0; r < 8; ++r)
        Ps[w][r + 8 * hi][j * 16 + lo] = (f16)sc2[j][r];

    v16h pf = join16(*(const v8h*)&Ps[w][lo][hi * 8],
                     *(const v8h*)&Ps[w][lo][16 + hi * 8]);

#pragma unroll
    for (int jj = 0; jj < 8; ++jj) {
      const f16* vrow = vbase + (size_t)(jj * 16 + lo) * Sc + k0 + hi * 8;
      v16h vf = join16(*(const v8h*)vrow, *(const v8h*)(vrow + 16));
      o[jj] = WMMA_F16(pf, vf, o[jj]);
    }
  }

  float invl[8];
#pragma unroll
  for (int r = 0; r < 8; ++r) invl[r] = 1.0f / l[r];
#pragma unroll
  for (int jj = 0; jj < 8; ++jj)
#pragma unroll
    for (int r = 0; r < 8; ++r) {
      int qpos = qb + r + 8 * hi;
      ao[(size_t)(b * Sc + qpos) * Dc + h * HDc + jj * 16 + lo] =
          (f16)(o[jj][r] * invl[r]);
    }
}

// ---------------------------------------------------------------------------
// launch
// ---------------------------------------------------------------------------
extern "C" void kernel_launch(void* const* d_in, const int* in_sizes, int n_in,
                              void* d_out, int out_size, void* d_ws, size_t ws_size,
                              hipStream_t stream) {
  (void)in_sizes; (void)n_in; (void)out_size; (void)ws_size;
  const float* x  = (const float*)d_in[0];
  const float* wq = (const float*)d_in[1];
  const float* wk = (const float*)d_in[2];
  const float* wv = (const float*)d_in[3];
  const float* wo = (const float*)d_in[4];
  const float* fc = (const float*)d_in[5];
  const float* fs = (const float*)d_in[6];
  // d_in[7] = mask (computed analytically), d_in[8] = start_pos (== 0)

  const size_t MB = (size_t)1 << 20;
  const int nx = Bc * Sc * Dc;   // 8M elements
  const int nw = Dc * Dc;        // 4M elements
  char* ws = (char*)d_ws;

  f16* xh  = (f16*)(ws + 0);        // 16 MB
  f16* wqh = (f16*)(ws + 16 * MB);  //  8 MB
  f16* wkh = (f16*)(ws + 24 * MB);  //  8 MB
  f16* wvh = (f16*)(ws + 32 * MB);  //  8 MB
  f16* woh = (f16*)(ws + 40 * MB);  //  8 MB
  f16* qh  = (f16*)(ws + 48 * MB);  // 16 MB
  f16* kh  = (f16*)(ws + 64 * MB);  // 16 MB
  f16* vh  = (f16*)(ws + 80 * MB);  // 16 MB
  f16* vt  = (f16*)(ws + 16 * MB);  // 16 MB (reuses wqh/wkh after projections)
  f16* aoh = (f16*)(ws + 0);        // 16 MB (reuses xh after projections)

  f32_to_f16_kernel<<<nx / 4 / 256, 256, 0, stream>>>(x, xh, nx / 4);
  f32_to_f16_kernel<<<nw / 4 / 256, 256, 0, stream>>>(wq, wqh, nw / 4);
  f32_to_f16_kernel<<<nw / 4 / 256, 256, 0, stream>>>(wk, wkh, nw / 4);
  f32_to_f16_kernel<<<nw / 4 / 256, 256, 0, stream>>>(wv, wvh, nw / 4);
  f32_to_f16_kernel<<<nw / 4 / 256, 256, 0, stream>>>(wo, woh, nw / 4);

  const int M = Bc * Sc;                           // 4096
  const int gemm_blocks = (M / 128) * (Dc / 128);  // 512
  gemm_xwT_kernel<f16><<<gemm_blocks, 256, 0, stream>>>(xh, wqh, qh, M, Dc, Dc);
  gemm_xwT_kernel<f16><<<gemm_blocks, 256, 0, stream>>>(xh, wkh, kh, M, Dc, Dc);
  gemm_xwT_kernel<f16><<<gemm_blocks, 256, 0, stream>>>(xh, wvh, vh, M, Dc, Dc);

  const int npairs = Bc * Sc * Hc * (HDc / 2);  // 4M
  rope_kernel<<<npairs / 256, 256, 0, stream>>>(qh, fc, fs);
  rope_kernel<<<npairs / 256, 256, 0, stream>>>(kh, fc, fs);

  dim3 tgrid(HDc / 64, Sc / 64, Bc * Hc);
  vtrans_kernel<<<tgrid, 256, 0, stream>>>(vh, vt);

  const int attn_blocks = Bc * Hc * (Sc / 64);  // 1024
  attn_kernel<<<attn_blocks, 128, 0, stream>>>(qh, kh, vt, aoh);

  gemm_xwT_kernel<float><<<gemm_blocks, 256, 0, stream>>>(aoh, woh, (float*)d_out, M, Dc, Dc);
}
